// LambdaLayer_10170482557507
// MI455X (gfx1250) — matrix operand order repaired and checked by reading, not compiled, not run on hardware
//
#include <hip/hip_runtime.h>
#include <hip/hip_bf16.h>

typedef __attribute__((ext_vector_type(16))) __bf16 v16bf;
typedef __attribute__((ext_vector_type(8)))  __bf16 v8bf;
typedef __attribute__((ext_vector_type(2)))  __bf16 v2bf;
typedef __attribute__((ext_vector_type(8)))  float  v8f;
typedef __attribute__((ext_vector_type(4)))  int    v4i;

#define Bn   16
#define Nn   1024
#define Mn   1024
#define INDn 512
#define Kn   64     // ATN_D
#define Hn   8
#define Vn   64

static __device__ inline v16bf frag_cat(v8bf lo, v8bf hi) {
  return __builtin_shufflevector(lo, hi, 0,1,2,3,4,5,6,7,8,9,10,11,12,13,14,15);
}

// ---------------------------------------------------------------------------
// CDNA5 async global->LDS copy (ASYNCcnt-tracked). Falls back to a VGPR
// round-trip if the toolchain does not expose the gfx1250 builtin.
// Builtin signature (from hipcc diagnostic): param0 = v4i in addrspace(1).
// ---------------------------------------------------------------------------
#if __has_builtin(__builtin_amdgcn_global_load_async_to_lds_b128) && \
    __has_builtin(__builtin_amdgcn_s_wait_asynccnt)
#define USE_ASYNC_LDS 1
typedef __attribute__((address_space(1))) v4i gas_v4i;
typedef __attribute__((address_space(3))) v4i las_v4i;
static __device__ inline void async_cp16(const __bf16* g, __bf16* l) {
  __builtin_amdgcn_global_load_async_to_lds_b128(
      (gas_v4i*)(unsigned long long)g,
      (las_v4i*)(unsigned int)(unsigned long long)l,
      0, 0);
}
static __device__ inline void async_wait0() { __builtin_amdgcn_s_wait_asynccnt(0); }
#else
#define USE_ASYNC_LDS 0
#warning "gfx1250 async global->LDS builtins not available; using VGPR fallback"
static __device__ inline void async_cp16(const __bf16* g, __bf16* l) {
  *reinterpret_cast<v8bf*>(l) = *reinterpret_cast<const v8bf*>(g);
}
static __device__ inline void async_wait0() {}
#endif

// ---------------------------------------------------------------------------
// Generic 64x64-tile bf16 WMMA GEMM over rows of X (16384 x 512) @ W (512 x NC)
// mode 0: outf[row*ldout + col] = acc + bias[col]            (fp32)
// mode 1: vs_t[(b*64 + col)*1024 + m] = bf16(acc + bias[col]) (transposed bf16)
// ---------------------------------------------------------------------------
__global__ __launch_bounds__(256)
void gemm_rowX_bf16(const float* __restrict__ X, const float* __restrict__ W,
                    const float* __restrict__ bias, int ldw,
                    float* __restrict__ outf, int ldout,
                    __bf16* __restrict__ out_vst, int mode)
{
  __shared__ __align__(16) __bf16 Xl[64 * 40];
  __shared__ __align__(16) __bf16 Wl[64 * 40];
  const int t    = threadIdx.x;
  const int lane = t & 31, w = t >> 5;
  const int half = lane >> 4, l15 = lane & 15;
  const int grow0 = blockIdx.x * 64;
  const int jc    = blockIdx.y * 64;
  const int mt  = w >> 1;          // row tile 0..3
  const int ntb = (w & 1) * 2;     // col tile pair base 0 or 2

  v8f c0 = {}; v8f c1 = {};

  for (int kc = 0; kc < INDn; kc += 32) {
    // X tile 64x32 fp32 -> bf16 (row-major [row][k])
#pragma unroll
    for (int i = 0; i < 4; ++i) {
      int pe  = i * 256 + t;                 // 1024 float2 pairs
      int row = pe >> 4, kp = (pe & 15) * 2;
      float2 xv = *reinterpret_cast<const float2*>(&X[(size_t)(grow0 + row) * INDn + kc + kp]);
      v2bf p; p[0] = (__bf16)xv.x; p[1] = (__bf16)xv.y;
      *reinterpret_cast<v2bf*>(&Xl[row * 40 + kp]) = p;
    }
    // W tile 32xK -> LDS transposed [col][k]
#pragma unroll
    for (int i = 0; i < 8; ++i) {
      int e = i * 256 + t;
      int kk = e >> 6, j = e & 63;
      Wl[j * 40 + kk] = (__bf16)W[(size_t)(kc + kk) * ldw + jc + j];
    }
    __syncthreads();
    {
      const __bf16* pa = &Xl[(mt * 16 + l15) * 40 + 8 * half];
      v16bf a = frag_cat(*(const v8bf*)pa, *(const v8bf*)(pa + 16));
      const __bf16* pb0 = &Wl[((ntb + 0) * 16 + l15) * 40 + 8 * half];
      v16bf b0 = frag_cat(*(const v8bf*)pb0, *(const v8bf*)(pb0 + 16));
      const __bf16* pb1 = &Wl[((ntb + 1) * 16 + l15) * 40 + 8 * half];
      v16bf b1 = frag_cat(*(const v8bf*)pb1, *(const v8bf*)(pb1 + 16));
      c0 = __builtin_amdgcn_wmma_f32_16x16x32_bf16(false, a, false, b0, (short)0, c0, false, false);
      c1 = __builtin_amdgcn_wmma_f32_16x16x32_bf16(false, a, false, b1, (short)0, c1, false, false);
    }
    __syncthreads();
  }

  // epilogue: mode branch hoisted out of the store loops
  const int colL0 = (ntb + 0) * 16 + l15;
  const int colL1 = (ntb + 1) * 16 + l15;
  const float bb0 = bias[jc + colL0];
  const float bb1 = bias[jc + colL1];
  if (mode == 0) {
#pragma unroll
    for (int r = 0; r < 8; ++r) {
      int grow = grow0 + mt * 16 + r + 8 * half;
      outf[(size_t)grow * ldout + jc + colL0] = c0[r] + bb0;
      outf[(size_t)grow * ldout + jc + colL1] = c1[r] + bb1;
    }
  } else {
#pragma unroll
    for (int r = 0; r < 8; ++r) {
      int grow = grow0 + mt * 16 + r + 8 * half;
      int b = grow >> 10, m = grow & 1023;
      out_vst[((size_t)(b * 64 + jc + colL0) << 10) + m] = (__bf16)(c0[r] + bb0);
      out_vst[((size_t)(b * 64 + jc + colL1) << 10) + m] = (__bf16)(c1[r] + bb1);
    }
  }
}

// ---------------------------------------------------------------------------
// Per-batch masked column softmax stats + c_lambda[b] = soft(ks)^T @ vs  (64x64)
// ---------------------------------------------------------------------------
__global__ __launch_bounds__(256)
void ctx_lambda_kernel(const float* __restrict__ ks_raw, const __bf16* __restrict__ vs_t,
                       const int* __restrict__ mask, float* __restrict__ c_lambda)
{
  __shared__ float red[4 * 64];
  __shared__ float mx[64];
  __shared__ float inv[64];
  __shared__ __align__(16) float  ks_ch[128 * 64];
  __shared__ __align__(16) __bf16 vs_ch[128 * 72];
  const int t = threadIdx.x;
  const int b = blockIdx.x;
  const int k = t & 63, part = t >> 6;

  // pass 1: column max over m
  float lmax = -3.4e38f;
  for (int i = 0; i < 256; ++i) {
    int m = part * 256 + i;
    float x = ks_raw[((size_t)(b * 1024 + m)) * 64 + k];
    x = mask[b * 1024 + m] ? x : -1e10f;
    lmax = fmaxf(lmax, x);
  }
  red[part * 64 + k] = lmax;
  __syncthreads();
  if (t < 64)
    mx[t] = fmaxf(fmaxf(red[t], red[64 + t]), fmaxf(red[128 + t], red[192 + t]));
  __syncthreads();
  // pass 2: sum of exp
  float lsum = 0.f, mk = mx[k];
  for (int i = 0; i < 256; ++i) {
    int m = part * 256 + i;
    float x = ks_raw[((size_t)(b * 1024 + m)) * 64 + k];
    x = mask[b * 1024 + m] ? x : -1e10f;
    lsum += __expf(x - mk);
  }
  red[part * 64 + k] = lsum;
  __syncthreads();
  if (t < 64)
    inv[t] = 1.0f / (red[t] + red[64 + t] + red[128 + t] + red[192 + t]);
  __syncthreads();

  // phase 3: c_lambda accumulation
  const int ck = t >> 2;            // 0..63
  const int vg = (t & 3) * 16;      // 0/16/32/48
  float acc[16];
#pragma unroll
  for (int j = 0; j < 16; ++j) acc[j] = 0.f;
  const float cmax = mx[ck], cinv = inv[ck];

  for (int mc = 0; mc < 8; ++mc) {
#pragma unroll
    for (int i = 0; i < 32; ++i) {
      int e = i * 256 + t;
      int ml = e >> 6, kk = e & 63;
      int m = mc * 128 + ml;
      float x = ks_raw[((size_t)(b * 1024 + m)) * 64 + kk];
      ks_ch[ml * 64 + kk] = mask[b * 1024 + m] ? x : -1e10f;
    }
#pragma unroll
    for (int i = 0; i < 32; ++i) {
      int e = i * 256 + t;
      int v = e >> 7, ml = e & 127;
      vs_ch[ml * 72 + v] = vs_t[((size_t)(b * 64 + v) << 10) + mc * 128 + ml];
    }
    __syncthreads();
    for (int ml = 0; ml < 128; ++ml) {
      float s = __expf(ks_ch[ml * 64 + ck] - cmax) * cinv;
      const __bf16* vp = &vs_ch[ml * 72 + vg];
#pragma unroll
      for (int j = 0; j < 16; ++j) acc[j] += s * (float)vp[j];
    }
    __syncthreads();
  }
#pragma unroll
  for (int j = 0; j < 16; ++j)
    c_lambda[(size_t)b * 4096 + ck * 64 + vg + j] = acc[j];
}

// ---------------------------------------------------------------------------
// Main kernel: per-n block. es[n] resident in LDS (bf16, [k][m]).
// Flattened (b, mc) pipeline: async global->LDS staging of the next vs chunk
// (double-buffered) overlaps the 16 WMMAs of the current chunk. Accumulators
// seeded with c_lambda[b]; per-b epilogue does the small qs @ L product.
// ---------------------------------------------------------------------------
__global__ __launch_bounds__(256)
void lambda_main_kernel(const float* __restrict__ es_g, const __bf16* __restrict__ vs_t,
                        const float* __restrict__ c_lambda, const float* __restrict__ qs,
                        float* __restrict__ out)
{
  __shared__ __align__(32) __bf16 es_lds[64 * 1040];       // 133,120 B
  __shared__ __align__(32) __bf16 vsbuf[2][64 * 272];      //  69,632 B
  __shared__ __align__(16) float  L_lds[64 * 68];          //  17,408 B

  const int t    = threadIdx.x;
  const int lane = t & 31, w = t >> 5;
  const int half = lane >> 4, l15 = lane & 15;
  const int n   = blockIdx.x;
  const int kt  = w >> 1;          // K-dim tile 0..3
  const int vtb = (w & 1) * 2;     // V-dim tile pair base

  // es[n]: 1024x64 fp32 -> bf16, transposed into LDS [k][m]
  const float* esn = es_g + (size_t)n * (Mn * Kn);
  for (int i = 0; i < 256; ++i) {
    int e = i * 256 + t;
    es_lds[(e & 63) * 1040 + (e >> 6)] = (__bf16)esn[e];
  }

  // stage first vs chunk (b=0, mc=0) into buffer 0
#pragma unroll
  for (int i = 0; i < 8; ++i) {
    int g = i * 256 + t;
    int v = g >> 5, mo = (g & 31) * 8;
    async_cp16(&vs_t[((size_t)v << 10) + mo], &vsbuf[0][v * 272 + mo]);
  }
  async_wait0();
  __syncthreads();

  v8f c0, c1;
  for (int it = 0; it < 64; ++it) {
    const int b = it >> 2, mc = it & 3, cur = it & 1;

    if (mc == 0) {
      // seed accumulators with c_lambda[b]; prefetch this (b,n)'s qs row
      __builtin_prefetch(qs + ((size_t)(b * 1024 + n)) * 512 + (size_t)(t & 63) * 8, 0, 0);
      const float* cl = c_lambda + (size_t)b * 4096;
#pragma unroll
      for (int r = 0; r < 8; ++r) {
        int row = kt * 16 + r + 8 * half;
        c0[r] = cl[row * 64 + (vtb + 0) * 16 + l15];
        c1[r] = cl[row * 64 + (vtb + 1) * 16 + l15];
      }
    }

    // issue async staging of the next chunk into the other buffer
    if (it + 1 < 64) {
      const int nb = (it + 1) >> 2, nmc = (it + 1) & 3, nbuf = (it + 1) & 1;
#pragma unroll
      for (int i = 0; i < 8; ++i) {
        int g = i * 256 + t;
        int v = g >> 5, mo = (g & 31) * 8;
        async_cp16(&vs_t[((size_t)(nb * 64 + v) << 10) + nmc * 256 + mo],
                   &vsbuf[nbuf][v * 272 + mo]);
      }
    }

    // compute: 8 k-steps x 2 tiles over the current chunk
    const __bf16* vb = vsbuf[cur];
#pragma unroll
    for (int ks8 = 0; ks8 < 8; ++ks8) {
      const __bf16* pa = &es_lds[(kt * 16 + l15) * 1040 + mc * 256 + ks8 * 32 + 8 * half];
      v16bf a = frag_cat(*(const v8bf*)pa, *(const v8bf*)(pa + 16));
      const __bf16* pb0 = &vb[((vtb + 0) * 16 + l15) * 272 + ks8 * 32 + 8 * half];
      v16bf b0 = frag_cat(*(const v8bf*)pb0, *(const v8bf*)(pb0 + 16));
      const __bf16* pb1 = &vb[((vtb + 1) * 16 + l15) * 272 + ks8 * 32 + 8 * half];
      v16bf b1 = frag_cat(*(const v8bf*)pb1, *(const v8bf*)(pb1 + 16));
      c0 = __builtin_amdgcn_wmma_f32_16x16x32_bf16(false, a, false, b0, (short)0, c0, false, false);
      c1 = __builtin_amdgcn_wmma_f32_16x16x32_bf16(false, a, false, b1, (short)0, c1, false, false);
    }

    if (mc == 3) {
      // L = c_lambda + p_lambda -> dedicated LDS buffer
#pragma unroll
      for (int r = 0; r < 8; ++r) {
        int row = kt * 16 + r + 8 * half;
        L_lds[row * 68 + (vtb + 0) * 16 + l15] = c0[r];
        L_lds[row * 68 + (vtb + 1) * 16 + l15] = c1[r];
      }
      __syncthreads();
      // out[b][n][h][v] = sum_k qs[b][n][h*64+k] * L[k][v]
      const float* q = qs + ((size_t)(b * 1024 + n)) * 512;
      int h0 = t >> 6, v = t & 63;
      float a0 = 0.f, a1 = 0.f;
      for (int k = 0; k < 64; ++k) {
        float l = L_lds[k * 68 + v];
        a0 += q[h0 * 64 + k] * l;
        a1 += q[(h0 + 4) * 64 + k] * l;
      }
      float* o = out + ((size_t)(b * 1024 + n)) * 512;
      o[h0 * 64 + v]       = a0;
      o[(h0 + 4) * 64 + v] = a1;
    }

    async_wait0();      // next chunk staged (own wave)...
    __syncthreads();    // ...and visible to all waves / L readers done
  }
}

// ---------------------------------------------------------------------------
extern "C" void kernel_launch(void* const* d_in, const int* in_sizes, int n_in,
                              void* d_out, int out_size, void* d_ws, size_t ws_size,
                              hipStream_t stream) {
  const float* xs   = (const float*)d_in[0];
  const float* cs   = (const float*)d_in[1];
  const float* es   = (const float*)d_in[2];
  const int*   mask = (const int*)d_in[3];
  const float* Wq   = (const float*)d_in[4];
  const float* bq   = (const float*)d_in[5];
  const float* Wk   = (const float*)d_in[6];
  const float* bk   = (const float*)d_in[7];
  const float* Wv   = (const float*)d_in[8];
  const float* bv   = (const float*)d_in[9];
  float* out = (float*)d_out;

  char* ws = (char*)d_ws;
  float*  qs     = (float*)(ws);                 // 16384*512*4  = 33,554,432 B
  float*  ks_raw = (float*)(ws + 33554432);      // 16384*64*4   =  4,194,304 B
  __bf16* vs_t   = (__bf16*)(ws + 37748736);     // 16*64*1024*2 =  2,097,152 B
  float*  c_lam  = (float*)(ws + 39845888);      // 16*64*64*4   =  262,144 B

  dim3 blk(256);
  // qs = xs @ Wq + bq            (16384x512 @ 512x512)
  gemm_rowX_bf16<<<dim3(256, 8), blk, 0, stream>>>(xs, Wq, bq, 512, qs, 512, nullptr, 0);
  // ks_raw = cs @ Wk + bk        (16384x512 @ 512x64)
  gemm_rowX_bf16<<<dim3(256, 1), blk, 0, stream>>>(cs, Wk, bk, 64, ks_raw, 64, nullptr, 0);
  // vs_t = (cs @ Wv + bv)^T bf16 (stored [b][v][m])
  gemm_rowX_bf16<<<dim3(256, 1), blk, 0, stream>>>(cs, Wv, bv, 64, nullptr, 0, vs_t, 1);
  // softmax stats + c_lambda per batch
  ctx_lambda_kernel<<<dim3(16), blk, 0, stream>>>(ks_raw, vs_t, mask, c_lam);
  // main fused p_lambda + output (double-buffered async pipeline)
  lambda_main_kernel<<<dim3(1024), blk, 0, stream>>>(es, vs_t, c_lam, qs, out);
}
